// RNAFoldTemplate_80169859547214
// MI455X (gfx1250) — matrix operand order repaired and checked by reading, not compiled, not run on hardware
//
#include <hip/hip_runtime.h>
#include <math.h>

// ---------------------------------------------------------------------------
// CDNA5 (gfx1250) Evoformer block. Dominant FLOPs (triangle einsums, pair
// projections, pair transition) run on v_wmma_f32_16x16x32_bf16. The big
// per-channel GEMMs are LDS-tiled with the Tensor Data Mover
// (tensor_load_to_lds + s_wait_tensorcnt, double-buffered). Small
// single-track ops (attention, node transition) are VALU kernels.
// ---------------------------------------------------------------------------

typedef __attribute__((ext_vector_type(16))) __bf16          v16bf;
typedef __attribute__((ext_vector_type(8)))  float           v8f;
typedef __attribute__((ext_vector_type(8)))  unsigned short  v8u16;
typedef __attribute__((ext_vector_type(16))) unsigned short  v16u16;
typedef __attribute__((ext_vector_type(4)))  unsigned int    v4u;
typedef __attribute__((ext_vector_type(8)))  int             v8i;
typedef __attribute__((ext_vector_type(4)))  int             v4i;

#define DEV __device__ __forceinline__

DEV unsigned short f2bf(float f) {
  unsigned int u = __float_as_uint(f);
  u += 0x7FFFu + ((u >> 16) & 1u);          // round-to-nearest-even
  return (unsigned short)(u >> 16);
}
DEV float bf2f(unsigned short h) { return __uint_as_float(((unsigned int)h) << 16); }
DEV float sigm(float x) { return 1.0f / (1.0f + __expf(-x)); }

DEV v16u16 combine8(v8u16 lo, v8u16 hi) {
  return __builtin_shufflevector(lo, hi, 0,1,2,3,4,5,6,7,8,9,10,11,12,13,14,15);
}
DEV v8f wmma_bf16(v16u16 a, v16u16 b, v8f c) {
  return __builtin_amdgcn_wmma_f32_16x16x32_bf16(
      false, __builtin_bit_cast(v16bf, a),
      false, __builtin_bit_cast(v16bf, b),
      (short)0, c, false, false);
}

// TDM: DMA a 64x64 bf16 tile out of a 512x512 row-major slice into LDS,
// padding each 128B row with 4 DWORDs -> 144B LDS row stride (bank-conflict
// free, 16B aligned for ds_load_b128 fragment reads).
DEV void tdm_load_tile64(unsigned lds_addr, const unsigned short* gptr) {
  unsigned long long ga = (unsigned long long)gptr;
  v4u g0 = { 1u,                              // count=1, user descriptor
             lds_addr,                        // lds_addr [63:32]
             (unsigned)ga,                    // global_addr [95:64]
             (unsigned)((ga >> 32) & 0x1FFFFFFu) | 0x80000000u }; // addr hi | type=2
  // g1: data_size=2B, pad_enable, pad_interval=32DW, pad_amount=4DW,
  //     tensor_dim0=512, tensor_dim1=512, tile=64x64, dim0_stride=512
  v8i g1 = { 0x07110000,                      // ds=1 | pad_en | intv=4 | amt=3
             (int)(512u << 16),               // tensor_dim0 lo
             (int)(512u << 16),               // dim0 hi=0 | tensor_dim1 lo
             (int)(64u << 16),                // dim1 hi=0 | tile_dim0=64
             64,                              // tile_dim1=64, tile_dim2=0
             512, 0, 0 };                     // tensor_dim0_stride=512
  v4i z4 = {0, 0, 0, 0};
#if defined(__clang_major__) && (__clang_major__ >= 23)
  v8i z8 = {0, 0, 0, 0, 0, 0, 0, 0};
  __builtin_amdgcn_tensor_load_to_lds(g0, g1, z4, z4, z8, 0);
#else
  __builtin_amdgcn_tensor_load_to_lds(g0, g1, z4, z4, 0);
#endif
}

// ---------------------------------------------------------------------------
// Small utility kernels
// ---------------------------------------------------------------------------

// Transpose + convert f32 (K,N) weight -> bf16 (N,K) so WMMA B-operand lanes
// read 16 contiguous K elements.
__global__ void wprep_kernel(const float* __restrict__ src,
                             unsigned short* __restrict__ dst, int K, int N) {
  int idx = blockIdx.x * blockDim.x + threadIdx.x;
  if (idx >= K * N) return;
  int k = idx / N, n = idx % N;
  dst[n * K + k] = f2bf(src[idx]);
}

// torsion gate tw[b,l] = sigmoid(torsion @ Wt + bt) for both triangle blocks
__global__ void torsion_kernel(const float* __restrict__ tor,
                               const float* Wt0, const float* bt0,
                               const float* Wt1, const float* bt1,
                               float* tw0, float* tw1) {
  int i = blockIdx.x * blockDim.x + threadIdx.x;
  if (i >= 1024) return;
  const float* t6 = tor + i * 6;
  float s0 = bt0[0], s1 = bt1[0];
  for (int k = 0; k < 6; ++k) { s0 += t6[k] * Wt0[k]; s1 += t6[k] * Wt1[k]; }
  tw0[i] = sigm(s0); tw1[i] = sigm(s1);
}

// row-wise LayerNorm (thread per row) for the single track
__global__ void ln_rows_kernel(const float* __restrict__ src,
                               const float* __restrict__ w, const float* __restrict__ b,
                               float* __restrict__ dst, int rows, int width) {
  int r = blockIdx.x * blockDim.x + threadIdx.x;
  if (r >= rows) return;
  const float* s = src + (size_t)r * width;
  float mu = 0.f;
  for (int i = 0; i < width; ++i) mu += s[i];
  mu /= width;
  float var = 0.f;
  for (int i = 0; i < width; ++i) { float d = s[i] - mu; var += d * d; }
  var /= width;
  float rstd = rsqrtf(var + 1e-5f);
  float* o = dst + (size_t)r * width;
  for (int i = 0; i < width; ++i) o[i] = (s[i] - mu) * rstd * w[i] + b[i];
}

// ---------------------------------------------------------------------------
// Attention (single track, small FLOPs -> VALU)
// ---------------------------------------------------------------------------

__global__ void qkvg_kernel(const float* __restrict__ z,
                            const float* Wq, const float* Wk, const float* Wv,
                            const float* Wg, const float* bg,
                            float* Q, float* K, float* V, float* G) {
  int idx = blockIdx.x * blockDim.x + threadIdx.x;   // 1024*128
  int row = idx >> 7, col = idx & 127;
  const float* zr = z + (size_t)row * 128;
  float q = 0.f, k = 0.f, v = 0.f, g = 0.f;
  for (int i = 0; i < 128; ++i) {
    float zi = zr[i];
    q += zi * Wq[i * 128 + col];
    k += zi * Wk[i * 128 + col];
    v += zi * Wv[i * 128 + col];
    g += zi * Wg[i * 128 + col];
  }
  Q[idx] = q; K[idx] = k; V[idx] = v; G[idx] = sigm(g + bg[col]);
}

// bias[b,h,i,j] = sum_p pair[b,i,j,p] * Wb[p,h]  (coalesced via LDS tile)
__global__ void __launch_bounds__(256) bias_kernel(const float* __restrict__ pair,
                                                   const float* __restrict__ Wb,
                                                   float* __restrict__ bias) {
  __shared__ float pf[64][64];
  int t = threadIdx.x;
  size_t pos0 = (size_t)blockIdx.x * 64;
  {
    int row = t >> 2, c0 = (t & 3) << 4;
    const float* src = pair + (pos0 + row) * 64 + c0;
    float* drow = &pf[row][c0];
    ((float4*)drow)[0] = ((const float4*)src)[0];
    ((float4*)drow)[1] = ((const float4*)src)[1];
    ((float4*)drow)[2] = ((const float4*)src)[2];
    ((float4*)drow)[3] = ((const float4*)src)[3];
  }
  __syncthreads();
  if (t < 64) {
    size_t pos = pos0 + t;
    int j = (int)(pos & 511);
    size_t bi = pos >> 9;
    int i = (int)(bi & 511);
    int bb = (int)(bi >> 9);
    float h0 = 0.f, h1 = 0.f, h2 = 0.f, h3 = 0.f;
    for (int p = 0; p < 64; ++p) {
      float pv = pf[t][p];
      h0 += pv * Wb[p * 4 + 0];
      h1 += pv * Wb[p * 4 + 1];
      h2 += pv * Wb[p * 4 + 2];
      h3 += pv * Wb[p * 4 + 3];
    }
    size_t ob = ((((size_t)bb * 4) * 512 + i) * 512) + j;
    size_t hs = (size_t)512 * 512;
    bias[ob] = h0; bias[ob + hs] = h1; bias[ob + 2 * hs] = h2; bias[ob + 3 * hs] = h3;
  }
}

// one wave32 per (b,h,i): scores, softmax, weighted V, gate
__global__ void __launch_bounds__(256) attn_kernel(const float* __restrict__ Q,
                                                   const float* __restrict__ K,
                                                   const float* __restrict__ V,
                                                   const float* __restrict__ G,
                                                   const float* __restrict__ bias,
                                                   float* __restrict__ attx) {
  __shared__ float att[8][512];
  int wv = threadIdx.x >> 5, lane = threadIdx.x & 31;
  int wid = blockIdx.x * 8 + wv;          // (bb,h,i)
  int i = wid & 511;
  int h = (wid >> 9) & 3;
  int bb = wid >> 11;
  size_t qoff = (((size_t)bb * 512 + i) * 128) + h * 32;
  float qv = Q[qoff + lane];
  const float* Kb = K + (size_t)bb * 512 * 128 + h * 32;
  const float* bi = bias + ((((size_t)bb * 4 + h) * 512) + i) * 512;
  const float sc = 0.17677669529663687f;  // 32^-0.5
  float s[16];
  for (int jc = 0; jc < 16; ++jc) {
    int j = jc * 32 + lane;
    const float* kr = Kb + (size_t)j * 128;
    float acc = 0.f;
    for (int d = 0; d < 32; ++d) acc += __shfl(qv, d, 32) * kr[d];
    s[jc] = acc * sc + bi[j];
  }
  float m = s[0];
  for (int jc = 1; jc < 16; ++jc) m = fmaxf(m, s[jc]);
  for (int o = 16; o >= 1; o >>= 1) m = fmaxf(m, __shfl_xor(m, o, 32));
  float Z = 0.f;
  for (int jc = 0; jc < 16; ++jc) { s[jc] = __expf(s[jc] - m); Z += s[jc]; }
  for (int o = 16; o >= 1; o >>= 1) Z += __shfl_xor(Z, o, 32);
  float rz = 1.f / Z;
  for (int jc = 0; jc < 16; ++jc) att[wv][jc * 32 + lane] = s[jc] * rz;
  __syncthreads();
  const float* Vb = V + (size_t)bb * 512 * 128 + h * 32 + lane;
  float acc = 0.f;
  for (int j = 0; j < 512; ++j) acc += att[wv][j] * Vb[(size_t)j * 128];
  attx[qoff + lane] = acc * G[qoff + lane];
}

__global__ void attnres_kernel(const float* __restrict__ single,
                               const float* __restrict__ attx,
                               const float* __restrict__ Wo, const float* __restrict__ bo,
                               float* __restrict__ out) {
  int idx = blockIdx.x * blockDim.x + threadIdx.x;   // 131072
  int row = idx >> 7, col = idx & 127;
  const float* ar = attx + (size_t)row * 128;
  float acc = bo[col];
  for (int k = 0; k < 128; ++k) acc += ar[k] * Wo[k * 128 + col];
  out[idx] = single[idx] + acc;
}

// NodeTransition MLP
__global__ void nth_kernel(const float* __restrict__ z, const float* __restrict__ W1,
                           const float* __restrict__ b1, float* __restrict__ h) {
  int idx = blockIdx.x * blockDim.x + threadIdx.x;   // 1024*512
  int row = idx >> 9, col = idx & 511;
  const float* zr = z + (size_t)row * 128;
  float acc = b1[col];
  for (int k = 0; k < 128; ++k) acc += zr[k] * W1[k * 512 + col];
  h[idx] = fmaxf(acc, 0.f);
}
__global__ void ntout_kernel(const float* __restrict__ sres, const float* __restrict__ h,
                             const float* __restrict__ W2, const float* __restrict__ b2,
                             float* __restrict__ out) {
  int idx = blockIdx.x * blockDim.x + threadIdx.x;   // 131072
  int row = idx >> 7, col = idx & 127;
  const float* hr = h + (size_t)row * 512;
  float acc = b2[col];
  for (int k = 0; k < 512; ++k) acc += hr[k] * W2[k * 128 + col];
  out[idx] = sres[idx] + acc;
}

// ---------------------------------------------------------------------------
// Triangle update: projection kernel (LN + 5 WMMA projections + gates + LDS
// transpose so a/b land in (B,P,L,L) bf16 with contiguous K for the GEMM).
// Workgroup covers rows (bb, x, y0..y0+63) of the pair tensor.
// ---------------------------------------------------------------------------
__global__ void __launch_bounds__(256) tri_proj_kernel(
    const float* __restrict__ pairw,
    const float* __restrict__ ln_w, const float* __restrict__ ln_b,
    const unsigned short* __restrict__ WlT, const unsigned short* __restrict__ WlgT,
    const float* __restrict__ blg,
    const unsigned short* __restrict__ WrT, const unsigned short* __restrict__ WrgT,
    const float* __restrict__ brg,
    const unsigned short* __restrict__ WogT, const float* __restrict__ bog,
    const float* __restrict__ tw, int tw_on_a,
    unsigned short* __restrict__ at, unsigned short* __restrict__ bt,
    unsigned short* __restrict__ gout) {
  __shared__ float zf[64][64];
  __shared__ unsigned short zbf[64][72];
  __shared__ unsigned short abuf[64][72];
  __shared__ unsigned short bbuf[64][72];

  int t = threadIdx.x;
  int blk = blockIdx.x;
  int y0 = (blk & 7) << 6;
  int x  = (blk >> 3) & 511;
  int bb = blk >> 12;
  size_t posbase = ((((size_t)bb << 9) + x) << 9) + y0;

  { // coalesced 64x64 f32 tile load
    int row = t >> 2, c0 = (t & 3) << 4;
    const float* src = pairw + (posbase + row) * 64 + c0;
    float* drow = &zf[row][c0];
    ((float4*)drow)[0] = ((const float4*)src)[0];
    ((float4*)drow)[1] = ((const float4*)src)[1];
    ((float4*)drow)[2] = ((const float4*)src)[2];
    ((float4*)drow)[3] = ((const float4*)src)[3];
  }
  __syncthreads();
  if (t < 64) { // LayerNorm over P=64, store bf16 A-operand tile
    float mu = 0.f;
    for (int c = 0; c < 64; ++c) mu += zf[t][c];
    mu *= (1.f / 64.f);
    float var = 0.f;
    for (int c = 0; c < 64; ++c) { float d = zf[t][c] - mu; var += d * d; }
    var *= (1.f / 64.f);
    float rstd = rsqrtf(var + 1e-5f);
    for (int c = 0; c < 64; ++c)
      zbf[t][c] = f2bf((zf[t][c] - mu) * rstd * ln_w[c] + ln_b[c]);
  }
  __syncthreads();

  int wv = t >> 5, lane = t & 31;
  int n16 = lane & 15, half = lane >> 4;
  float twv = tw[(bb << 9) + x];
  float twa = tw_on_a ? twv : 1.f;
  float twb = tw_on_a ? 1.f : twv;

  for (int s = wv; s < 16; s += 8) {
    int mt = s >> 2, nt = s & 3;
    v8f accl = {}, acclg = {}, accr = {}, accrg = {}, accog = {};
    const unsigned short* zr = &zbf[mt * 16 + n16][half * 16];
    int wb = (nt * 16 + n16) * 64 + half * 16;
#pragma unroll
    for (int k0 = 0; k0 < 64; k0 += 32) {
      v16u16 A = combine8(*(const v8u16*)(zr + k0), *(const v8u16*)(zr + k0 + 8));
      accl  = wmma_bf16(A, *(const v16u16*)(WlT  + wb + k0), accl);
      acclg = wmma_bf16(A, *(const v16u16*)(WlgT + wb + k0), acclg);
      accr  = wmma_bf16(A, *(const v16u16*)(WrT  + wb + k0), accr);
      accrg = wmma_bf16(A, *(const v16u16*)(WrgT + wb + k0), accrg);
      accog = wmma_bf16(A, *(const v16u16*)(WogT + wb + k0), accog);
    }
    int nn = nt * 16 + n16;
    float blgn = blg[nn], brgn = brg[nn], bogn = bog[nn];
#pragma unroll
    for (int r = 0; r < 8; ++r) {
      int row = mt * 16 + half * 8 + r;
      abuf[row][nn] = f2bf(accl[r] * sigm(acclg[r] + blgn) * twa);
      bbuf[row][nn] = f2bf(accr[r] * sigm(accrg[r] + brgn) * twb);
      gout[(posbase + row) * 64 + nn] = f2bf(sigm(accog[r] + bogn));
    }
  }
  __syncthreads();

  { // transposed coalesced writeout: (row=y, col=d) -> at[bb][d][x][y]
    int d = t >> 2, yc = (t & 3) << 4;
    size_t ob = ((((size_t)bb * 64 + d) * 512) + x) * 512 + y0 + yc;
    unsigned int w[8];
#pragma unroll
    for (int i = 0; i < 8; ++i)
      w[i] = (unsigned)abuf[yc + 2 * i][d] | ((unsigned)abuf[yc + 2 * i + 1][d] << 16);
    uint4 u0 = {w[0], w[1], w[2], w[3]}, u1 = {w[4], w[5], w[6], w[7]};
    *(uint4*)(at + ob) = u0; *(uint4*)(at + ob + 8) = u1;
#pragma unroll
    for (int i = 0; i < 8; ++i)
      w[i] = (unsigned)bbuf[yc + 2 * i][d] | ((unsigned)bbuf[yc + 2 * i + 1][d] << 16);
    uint4 v0 = {w[0], w[1], w[2], w[3]}, v1 = {w[4], w[5], w[6], w[7]};
    *(uint4*)(bt + ob) = v0; *(uint4*)(bt + ob + 8) = v1;
  }
}

// batched 512x512 bf16 transpose (64x64 tiles) for the incoming triangle
__global__ void __launch_bounds__(256) tr64_kernel(const unsigned short* __restrict__ in,
                                                   unsigned short* __restrict__ out) {
  __shared__ unsigned short tile[64][72];
  int t = threadIdx.x;
  int blk = blockIdx.x;                 // 128 slices * 8 * 8
  int tx = blk & 7, ty = (blk >> 3) & 7, sl = blk >> 6;
  size_t base = (size_t)sl * 512 * 512;
  int row = t >> 2, c0 = (t & 3) << 4;
  const unsigned short* src = in + base + (size_t)(ty * 64 + row) * 512 + tx * 64 + c0;
  *(v8u16*)&tile[row][c0]     = *(const v8u16*)src;
  *(v8u16*)&tile[row][c0 + 8] = *(const v8u16*)(src + 8);
  __syncthreads();
  unsigned int w[8];
#pragma unroll
  for (int i = 0; i < 8; ++i)
    w[i] = (unsigned)tile[c0 + 2 * i][row] | ((unsigned)tile[c0 + 2 * i + 1][row] << 16);
  unsigned short* dst = out + base + (size_t)(tx * 64 + row) * 512 + ty * 64 + c0;
  uint4 u0 = {w[0], w[1], w[2], w[3]}, u1 = {w[4], w[5], w[6], w[7]};
  *(uint4*)dst = u0; *(uint4*)(dst + 8) = u1;
}

// ---------------------------------------------------------------------------
// per-(b,d) 512x512x512 GEMM: C[i,j] = sum_k A[i,k]*B[j,k].
// Workgroup computes a 64x64 tile; A/B 64x64 k-panels are DMA'd into LDS by
// the Tensor Data Mover (double-buffered, padded rows -> conflict-free
// ds_load_b128 fragment reads). 8 waves x two 16x16 subtiles sharing the A
// fragment.
// ---------------------------------------------------------------------------
__global__ void __launch_bounds__(256) tri_gemm_kernel(const unsigned short* __restrict__ A,
                                                       const unsigned short* __restrict__ Bm,
                                                       float* __restrict__ X, float scale) {
  __shared__ unsigned short Atile[2][64 * 72];
  __shared__ unsigned short Btile[2][64 * 72];
  int t = threadIdx.x;
  int blk = blockIdx.x;                 // 128 slices * 8 * 8
  int j0 = (blk & 7) << 6;
  int i0 = ((blk >> 3) & 7) << 6;
  int bd = blk >> 6;
  const unsigned short* Ab = A  + (size_t)bd * 262144 + (size_t)i0 * 512;
  const unsigned short* Bb = Bm + (size_t)bd * 262144 + (size_t)j0 * 512;
  unsigned ldsA0 = (unsigned)(unsigned long long)(void*)&Atile[0][0];
  unsigned ldsA1 = (unsigned)(unsigned long long)(void*)&Atile[1][0];
  unsigned ldsB0 = (unsigned)(unsigned long long)(void*)&Btile[0][0];
  unsigned ldsB1 = (unsigned)(unsigned long long)(void*)&Btile[1][0];

  int wv = t >> 5, lane = t & 31;
  int n16 = lane & 15, half = lane >> 4;
  int mt = wv >> 1, nt0 = (wv & 1) << 1;
  v8f acc0 = {}, acc1 = {};

  if (t < 32) {  // wave 0 drives the TDM
    tdm_load_tile64(ldsA0, Ab);
    tdm_load_tile64(ldsB0, Bb);
  }
  int buf = 0;
  for (int kc = 0; kc < 512; kc += 64) {
    if (t < 32) {
      if (kc + 64 < 512) {
        tdm_load_tile64(buf ? ldsA0 : ldsA1, Ab + kc + 64);
        tdm_load_tile64(buf ? ldsB0 : ldsB1, Bb + kc + 64);
        __builtin_amdgcn_s_wait_tensorcnt(2);   // current buffer's 2 loads done
      } else {
        __builtin_amdgcn_s_wait_tensorcnt(0);
      }
    }
    __syncthreads();
    const unsigned short* At_ = &Atile[buf][0];
    const unsigned short* Bt_ = &Btile[buf][0];
#pragma unroll
    for (int k0 = 0; k0 < 64; k0 += 32) {
      const unsigned short* ar = At_ + (mt * 16 + n16) * 72 + half * 16 + k0;
      v16u16 av = combine8(*(const v8u16*)ar, *(const v8u16*)(ar + 8));
      const unsigned short* br0 = Bt_ + (nt0 * 16 + n16) * 72 + half * 16 + k0;
      const unsigned short* br1 = Bt_ + ((nt0 + 1) * 16 + n16) * 72 + half * 16 + k0;
      acc0 = wmma_bf16(av, *(const v16u16*)br0, acc0);
      acc1 = wmma_bf16(av, *(const v16u16*)br1, acc1);
    }
    __syncthreads();   // protect buffer before next DMA overwrites it
    buf ^= 1;
  }
  float* xr = X + (size_t)bd * 262144 + (size_t)(i0 + mt * 16 + half * 8) * 512 + j0;
#pragma unroll
  for (int r = 0; r < 8; ++r) {
    xr[(size_t)r * 512 + nt0 * 16 + n16]       = acc0[r] * scale;
    xr[(size_t)r * 512 + (nt0 + 1) * 16 + n16] = acc1[r] * scale;
  }
}

// LN(x) @ Wop + bop, gated, residual into pair (in-place)
__global__ void __launch_bounds__(256) tri_out_kernel(
    const float* __restrict__ xt, const unsigned short* __restrict__ gbuf,
    const float* __restrict__ lno_w, const float* __restrict__ lno_b,
    const unsigned short* __restrict__ WopT, const float* __restrict__ bop,
    float* __restrict__ pairw) {
  __shared__ float zf[64][64];            // [j][d]
  __shared__ unsigned short zbf[64][72];
  int t = threadIdx.x;
  int blk = blockIdx.x;
  int j0 = (blk & 7) << 6;
  int x  = (blk >> 3) & 511;
  int bb = blk >> 12;
  size_t posbase = ((((size_t)bb << 9) + x) << 9) + j0;

  { // gather x_t rows (d-major) into LDS transposed
    int d = t >> 2, jc = (t & 3) << 4;
    const float* src = xt + ((((size_t)bb * 64 + d) * 512) + x) * 512 + j0 + jc;
#pragma unroll
    for (int i = 0; i < 16; i += 4) {
      float4 v = *(const float4*)(src + i);
      zf[jc + i + 0][d] = v.x; zf[jc + i + 1][d] = v.y;
      zf[jc + i + 2][d] = v.z; zf[jc + i + 3][d] = v.w;
    }
  }
  __syncthreads();
  if (t < 64) {
    float mu = 0.f;
    for (int c = 0; c < 64; ++c) mu += zf[t][c];
    mu *= (1.f / 64.f);
    float var = 0.f;
    for (int c = 0; c < 64; ++c) { float d = zf[t][c] - mu; var += d * d; }
    var *= (1.f / 64.f);
    float rstd = rsqrtf(var + 1e-5f);
    for (int c = 0; c < 64; ++c)
      zbf[t][c] = f2bf((zf[t][c] - mu) * rstd * lno_w[c] + lno_b[c]);
  }
  __syncthreads();

  int wv = t >> 5, lane = t & 31;
  int n16 = lane & 15, half = lane >> 4;
  for (int s = wv; s < 16; s += 8) {
    int mt = s >> 2, nt = s & 3;
    v8f acc = {};
    const unsigned short* zr = &zbf[mt * 16 + n16][half * 16];
    int wb = (nt * 16 + n16) * 64 + half * 16;
#pragma unroll
    for (int k0 = 0; k0 < 64; k0 += 32) {
      v16u16 A = combine8(*(const v8u16*)(zr + k0), *(const v8u16*)(zr + k0 + 8));
      acc = wmma_bf16(A, *(const v16u16*)(WopT + wb + k0), acc);
    }
    int nn = nt * 16 + n16;
    float bopn = bop[nn];
#pragma unroll
    for (int r = 0; r < 8; ++r) {
      int row = mt * 16 + half * 8 + r;
      size_t pi = (posbase + row) * 64 + nn;
      pairw[pi] += (acc[r] + bopn) * bf2f(gbuf[pi]);
    }
  }
}

// PairTransition: LN -> (x@W1+b1).relu -> @W2+b2 -> +residual, fused via LDS
__global__ void __launch_bounds__(256) pt_kernel(
    const float* __restrict__ pairw,
    const float* __restrict__ ln_w, const float* __restrict__ ln_b,
    const unsigned short* __restrict__ W1T, const float* __restrict__ b1,
    const unsigned short* __restrict__ W2T, const float* __restrict__ b2,
    float* __restrict__ outp) {
  __shared__ float pf[64][64];
  __shared__ unsigned short zbf[64][72];
  __shared__ unsigned short hbuf[64][264];
  int t = threadIdx.x;
  size_t pos0 = (size_t)blockIdx.x * 64;
  {
    int row = t >> 2, c0 = (t & 3) << 4;
    const float* src = pairw + (pos0 + row) * 64 + c0;
    float* drow = &pf[row][c0];
    ((float4*)drow)[0] = ((const float4*)src)[0];
    ((float4*)drow)[1] = ((const float4*)src)[1];
    ((float4*)drow)[2] = ((const float4*)src)[2];
    ((float4*)drow)[3] = ((const float4*)src)[3];
  }
  __syncthreads();
  if (t < 64) {
    float mu = 0.f;
    for (int c = 0; c < 64; ++c) mu += pf[t][c];
    mu *= (1.f / 64.f);
    float var = 0.f;
    for (int c = 0; c < 64; ++c) { float d = pf[t][c] - mu; var += d * d; }
    var *= (1.f / 64.f);
    float rstd = rsqrtf(var + 1e-5f);
    for (int c = 0; c < 64; ++c)
      zbf[t][c] = f2bf((pf[t][c] - mu) * rstd * ln_w[c] + ln_b[c]);
  }
  __syncthreads();

  int wv = t >> 5, lane = t & 31;
  int n16 = lane & 15, half = lane >> 4;
  for (int s = wv; s < 64; s += 8) {       // h = relu(z@W1 + b1), N=256
    int mt = s >> 4, nt = s & 15;
    v8f acc = {};
    const unsigned short* zr = &zbf[mt * 16 + n16][half * 16];
    int wb = (nt * 16 + n16) * 64 + half * 16;
#pragma unroll
    for (int k0 = 0; k0 < 64; k0 += 32) {
      v16u16 A = combine8(*(const v8u16*)(zr + k0), *(const v8u16*)(zr + k0 + 8));
      acc = wmma_bf16(A, *(const v16u16*)(W1T + wb + k0), acc);
    }
    int nn = nt * 16 + n16;
    float b1n = b1[nn];
#pragma unroll
    for (int r = 0; r < 8; ++r) {
      int row = mt * 16 + half * 8 + r;
      hbuf[row][nn] = f2bf(fmaxf(acc[r] + b1n, 0.f));
    }
  }
  __syncthreads();
  for (int s = wv; s < 16; s += 8) {       // out = h@W2 + b2 + residual
    int mt = s >> 2, nt = s & 3;
    v8f acc = {};
    const unsigned short* hr = &hbuf[mt * 16 + n16][half * 16];
    int wb = (nt * 16 + n16) * 256 + half * 16;
#pragma unroll
    for (int k0 = 0; k0 < 256; k0 += 32) {
      v16u16 A = combine8(*(const v8u16*)(hr + k0), *(const v8u16*)(hr + k0 + 8));
      acc = wmma_bf16(A, *(const v16u16*)(W2T + wb + k0), acc);
    }
    int nn = nt * 16 + n16;
    float b2n = b2[nn];
#pragma unroll
    for (int r = 0; r < 8; ++r) {
      int row = mt * 16 + half * 8 + r;
      outp[(pos0 + row) * 64 + nn] = acc[r] + b2n + pf[row][nn];
    }
  }
}

// ---------------------------------------------------------------------------
// Host side
// ---------------------------------------------------------------------------
extern "C" void kernel_launch(void* const* d_in, const int* in_sizes, int n_in,
                              void* d_out, int out_size, void* d_ws, size_t ws_size,
                              hipStream_t stream) {
  (void)in_sizes; (void)n_in; (void)out_size; (void)ws_size;
  const float* single  = (const float*)d_in[0];
  const float* pair    = (const float*)d_in[1];
  const float* torsion = (const float*)d_in[2];
  const float* ra_ln_w = (const float*)d_in[3];
  const float* ra_ln_b = (const float*)d_in[4];
  const float* ra_Wq   = (const float*)d_in[5];
  const float* ra_Wk   = (const float*)d_in[6];
  const float* ra_Wv   = (const float*)d_in[7];
  const float* ra_Wb   = (const float*)d_in[8];
  const float* ra_Wg   = (const float*)d_in[9];
  const float* ra_bg   = (const float*)d_in[10];
  const float* ra_Wo   = (const float*)d_in[11];
  const float* ra_bo   = (const float*)d_in[12];
  const float* nt_ln_w = (const float*)d_in[13];
  const float* nt_ln_b = (const float*)d_in[14];
  const float* nt_W1   = (const float*)d_in[15];
  const float* nt_b1   = (const float*)d_in[16];
  const float* nt_W2   = (const float*)d_in[17];
  const float* nt_b2   = (const float*)d_in[18];
  const float* pt_ln_w = (const float*)d_in[19];
  const float* pt_ln_b = (const float*)d_in[20];
  const float* pt_W1   = (const float*)d_in[21];
  const float* pt_b1   = (const float*)d_in[22];
  const float* pt_W2   = (const float*)d_in[23];
  const float* pt_b2   = (const float*)d_in[24];
  const float* tri_in[2][16];
  for (int p = 0; p < 2; ++p)
    for (int k = 0; k < 16; ++k) tri_in[p][k] = (const float*)d_in[25 + p * 16 + k];
  // tri_in order: ln_w, ln_b, Wl, Wr, Wlg, blg, Wrg, brg, Wog, bog, Wop, bop,
  //               lno_w, lno_b, Wt, bt

  const size_t SZ_PAIR = 134217728, SZ_BF = 67108864, SZ_SD = 524288;
  char* ws = (char*)d_ws;
  size_t o = 0;
  auto alloc = [&](size_t bytes) { size_t r = o; o += (bytes + 255) & ~(size_t)255; return r; };
  float* pairw   = (float*)(ws + alloc(SZ_PAIR));
  float* XT      = (float*)(ws + alloc(SZ_PAIR));
  unsigned short* AT  = (unsigned short*)(ws + alloc(SZ_BF));
  unsigned short* BT  = (unsigned short*)(ws + alloc(SZ_BF));
  unsigned short* AT2 = (unsigned short*)(ws + alloc(SZ_BF));
  unsigned short* BT2 = (unsigned short*)(ws + alloc(SZ_BF));
  unsigned short* Gb  = (unsigned short*)(ws + alloc(SZ_BF));
  float* bias    = (float*)(ws + alloc(8388608));
  float* Z1      = (float*)(ws + alloc(SZ_SD));
  float* Qb      = (float*)(ws + alloc(SZ_SD));
  float* Kb      = (float*)(ws + alloc(SZ_SD));
  float* Vb      = (float*)(ws + alloc(SZ_SD));
  float* Gg      = (float*)(ws + alloc(SZ_SD));
  float* ATTX    = (float*)(ws + alloc(SZ_SD));
  float* SWS     = (float*)(ws + alloc(SZ_SD));
  float* Z2      = (float*)(ws + alloc(SZ_SD));
  float* Hb      = (float*)(ws + alloc(2097152));
  float* TW0     = (float*)(ws + alloc(4096));
  float* TW1     = (float*)(ws + alloc(4096));
  unsigned short* WT_TRI[2][6];   // Wl, Wr, Wlg, Wrg, Wog, Wop (64x64 each)
  for (int p = 0; p < 2; ++p)
    for (int k = 0; k < 6; ++k) WT_TRI[p][k] = (unsigned short*)(ws + alloc(8192));
  unsigned short* WT_PT1 = (unsigned short*)(ws + alloc(32768));
  unsigned short* WT_PT2 = (unsigned short*)(ws + alloc(32768));

  float* out_single = (float*)d_out;
  float* out_pair   = (float*)d_out + 131072;
  const float xscale = 1.0f / (sqrtf(512.0f) + 1e-8f);

  // --- weight prep (bf16, transposed) ---
  const int wsel[6] = {2, 3, 4, 6, 8, 10};  // Wl, Wr, Wlg, Wrg, Wog, Wop
  for (int p = 0; p < 2; ++p)
    for (int k = 0; k < 6; ++k)
      wprep_kernel<<<16, 256, 0, stream>>>(tri_in[p][wsel[k]], WT_TRI[p][k], 64, 64);
  wprep_kernel<<<64, 256, 0, stream>>>(pt_W1, WT_PT1, 64, 256);
  wprep_kernel<<<64, 256, 0, stream>>>(pt_W2, WT_PT2, 256, 64);
  torsion_kernel<<<4, 256, 0, stream>>>(torsion, tri_in[0][14], tri_in[0][15],
                                        tri_in[1][14], tri_in[1][15], TW0, TW1);

  // --- single track: attention + node transition ---
  ln_rows_kernel<<<4, 256, 0, stream>>>(single, ra_ln_w, ra_ln_b, Z1, 1024, 128);
  qkvg_kernel<<<512, 256, 0, stream>>>(Z1, ra_Wq, ra_Wk, ra_Wv, ra_Wg, ra_bg, Qb, Kb, Vb, Gg);
  bias_kernel<<<8192, 256, 0, stream>>>(pair, ra_Wb, bias);
  attn_kernel<<<512, 256, 0, stream>>>(Qb, Kb, Vb, Gg, bias, ATTX);
  attnres_kernel<<<512, 256, 0, stream>>>(single, ATTX, ra_Wo, ra_bo, SWS);
  ln_rows_kernel<<<4, 256, 0, stream>>>(SWS, nt_ln_w, nt_ln_b, Z2, 1024, 128);
  nth_kernel<<<2048, 256, 0, stream>>>(Z2, nt_W1, nt_b1, Hb);
  ntout_kernel<<<512, 256, 0, stream>>>(SWS, Hb, nt_W2, nt_b2, out_single);

  // --- pair track ---
  hipMemcpyAsync(pairw, pair, SZ_PAIR, hipMemcpyDeviceToDevice, stream);
  for (int p = 0; p < 2; ++p) {
    const float* const* ti = tri_in[p];
    float* tw = p == 0 ? TW0 : TW1;
    tri_proj_kernel<<<8192, 256, 0, stream>>>(
        pairw, ti[0], ti[1],
        WT_TRI[p][0], WT_TRI[p][2], ti[5],
        WT_TRI[p][1], WT_TRI[p][3], ti[7],
        WT_TRI[p][4], ti[9],
        tw, p == 0 ? 1 : 0, AT, BT, Gb);
    const unsigned short* ga = AT;
    const unsigned short* gb = BT;
    if (p == 1) {  // incoming: contraction over first L index -> transpose both
      tr64_kernel<<<8192, 256, 0, stream>>>(AT, AT2);
      tr64_kernel<<<8192, 256, 0, stream>>>(BT, BT2);
      ga = AT2; gb = BT2;
    }
    tri_gemm_kernel<<<8192, 256, 0, stream>>>(ga, gb, XT, xscale);
    tri_out_kernel<<<8192, 256, 0, stream>>>(XT, Gb, ti[12], ti[13],
                                             WT_TRI[p][5], ti[11], pairw);
  }
  pt_kernel<<<8192, 256, 0, stream>>>(pairw, pt_ln_w, pt_ln_b,
                                      WT_PT1, pt_b1, WT_PT2, pt_b2, out_pair);
}
